// TextQueryCrossAttention_64338610094967
// MI455X (gfx1250) — matrix operand order repaired and checked by reading, not compiled, not run on hardware
//
#include <hip/hip_runtime.h>

// ---------------------------------------------------------------- constants
constexpr int B_  = 2;
constexpr int S_  = 4096;
constexpr int NQ_ = 576;
constexpr int H_  = 32;
constexpr int HD_ = 128;
constexpr int HID_ = 4096;

// ---------------------------------------------------------------- types
typedef __attribute__((ext_vector_type(16))) __bf16 v16bf;
typedef __attribute__((ext_vector_type(8)))  float  v8f;

union ABFrag { v16bf v; uint4 q[2]; };
union CFrag  { v8f   v; float f[8]; };

__device__ __forceinline__ unsigned int f2bf_bits(float x) {
  unsigned int u = __float_as_uint(x);
  u += 0x7FFFu + ((u >> 16) & 1u);          // round-to-nearest-even
  return u >> 16;
}
__device__ __forceinline__ unsigned short f2bf(float x) {
  return (unsigned short)f2bf_bits(x);
}

// ---------------------------------------------------------------- fp32 -> bf16
// 8 elements per thread: two float4 loads -> one uint4 (8 x bf16) store.
__global__ void cvt_f32_bf16_v8(const float4* __restrict__ in,
                                uint4* __restrict__ out, size_t n8) {
  size_t i = (size_t)blockIdx.x * blockDim.x + threadIdx.x;
  size_t stride = (size_t)gridDim.x * blockDim.x;
  for (; i < n8; i += stride) {
    float4 a = in[2 * i];
    float4 b = in[2 * i + 1];
    uint4 o;
    o.x = f2bf_bits(a.x) | (f2bf_bits(a.y) << 16);
    o.y = f2bf_bits(a.z) | (f2bf_bits(a.w) << 16);
    o.z = f2bf_bits(b.x) | (f2bf_bits(b.y) << 16);
    o.w = f2bf_bits(b.z) | (f2bf_bits(b.w) << 16);
    out[i] = o;
  }
}

// ---------------------------------------------------------------- V transpose
// V: (B*NQ, HID) bf16  ->  Vt: (B, H, HD, NQ) bf16 (contiguous along NQ)
__global__ void transpose_v_kernel(const unsigned short* __restrict__ V,
                                   unsigned short* __restrict__ Vt) {
  const size_t total = (size_t)B_ * H_ * HD_ * NQ_;
  size_t idx = (size_t)blockIdx.x * blockDim.x + threadIdx.x;
  if (idx >= total) return;
  int q = (int)(idx % NQ_); size_t r = idx / NQ_;
  int d = (int)(r % HD_);   r /= HD_;
  int h = (int)(r % H_);    int b = (int)(r / H_);
  Vt[idx] = V[((size_t)b * NQ_ + q) * HID_ + (size_t)h * HD_ + d];
}

// ---------------------------------------------------------------- GEMM  C = A * W^T
// A: (M,K) bf16 row-major, W: (N,K) bf16 row-major.
// Block: 256 threads = 8 waves (2x4). Wave tile 32(M) x 64(N) = 2x4 WMMA tiles.
// Block tile 64(M) x 256(N). 8 WMMAs per 12 b128 loads per K-step.
template <bool OUT_BF16>
__global__ __launch_bounds__(256) void gemm_xwt(
    const unsigned short* __restrict__ A,
    const unsigned short* __restrict__ W,
    void* __restrict__ Cout, int M, int N, int K) {
  const int lane = threadIdx.x & 31;
  const int wave = threadIdx.x >> 5;
  const int wm = wave >> 2, wn = wave & 3;
  const int m0 = blockIdx.y * 64 + wm * 32;
  const int n0 = blockIdx.x * 256 + wn * 64;
  const int row = lane & 15, hi = lane >> 4;

  // A fragment (16x32): lane holds row (lane&15), 8-elem chunks at k+8*hi, k+8*hi+16
  const unsigned short* ap0 = A + (size_t)(m0 + row) * K + hi * 8;
  const unsigned short* ap1 = ap0 + (size_t)16 * K;
  // B fragment (32x16 = W^T tile): lane holds col n (row n of W), 8-elem chunks at k+8*hi, k+8*hi+16
  const unsigned short* bp0 = W + (size_t)(n0 + row) * K + hi * 8;
  const unsigned short* bp1 = bp0 + (size_t)16 * K;
  const unsigned short* bp2 = bp0 + (size_t)32 * K;
  const unsigned short* bp3 = bp0 + (size_t)48 * K;

  CFrag acc[2][4];
  for (int t = 0; t < 2; ++t)
    for (int u = 0; u < 4; ++u)
      for (int i = 0; i < 8; ++i) acc[t][u].f[i] = 0.0f;

  for (int k0 = 0; k0 < K; k0 += 32) {
    ABFrag a[2], b[4];
    a[0].q[0] = *reinterpret_cast<const uint4*>(ap0 + k0);
    a[0].q[1] = *reinterpret_cast<const uint4*>(ap0 + k0 + 16);
    a[1].q[0] = *reinterpret_cast<const uint4*>(ap1 + k0);
    a[1].q[1] = *reinterpret_cast<const uint4*>(ap1 + k0 + 16);
    b[0].q[0] = *reinterpret_cast<const uint4*>(bp0 + k0);
    b[0].q[1] = *reinterpret_cast<const uint4*>(bp0 + k0 + 16);
    b[1].q[0] = *reinterpret_cast<const uint4*>(bp1 + k0);
    b[1].q[1] = *reinterpret_cast<const uint4*>(bp1 + k0 + 16);
    b[2].q[0] = *reinterpret_cast<const uint4*>(bp2 + k0);
    b[2].q[1] = *reinterpret_cast<const uint4*>(bp2 + k0 + 16);
    b[3].q[0] = *reinterpret_cast<const uint4*>(bp3 + k0);
    b[3].q[1] = *reinterpret_cast<const uint4*>(bp3 + k0 + 16);

    // prefetch the streaming activation operand 8 K-steps (256 elements) ahead
    __builtin_prefetch(ap0 + k0 + 256, 0, 0);
    __builtin_prefetch(ap1 + k0 + 256, 0, 0);

    for (int t = 0; t < 2; ++t) {
      acc[t][0].v = __builtin_amdgcn_wmma_f32_16x16x32_bf16(false, a[t].v, false, b[0].v, (short)0, acc[t][0].v, false, false);
      acc[t][1].v = __builtin_amdgcn_wmma_f32_16x16x32_bf16(false, a[t].v, false, b[1].v, (short)0, acc[t][1].v, false, false);
      acc[t][2].v = __builtin_amdgcn_wmma_f32_16x16x32_bf16(false, a[t].v, false, b[2].v, (short)0, acc[t][2].v, false, false);
      acc[t][3].v = __builtin_amdgcn_wmma_f32_16x16x32_bf16(false, a[t].v, false, b[3].v, (short)0, acc[t][3].v, false, false);
    }
  }

  // C layout: VGPR i -> M = i + 8*(lane>=16), N = lane&15
  for (int t = 0; t < 2; ++t)
    for (int u = 0; u < 4; ++u) {
      const int mb = m0 + t * 16 + 8 * hi;
      const int nb = n0 + u * 16 + row;
      if (OUT_BF16) {
        unsigned short* Cp = (unsigned short*)Cout;
        for (int i = 0; i < 8; ++i)
          Cp[(size_t)(mb + i) * N + nb] = f2bf(acc[t][u].f[i]);
      } else {
        float* Cp = (float*)Cout;
        for (int i = 0; i < 8; ++i)
          Cp[(size_t)(mb + i) * N + nb] = acc[t][u].f[i];
      }
    }
}

// ---------------------------------------------------------------- attention
// One wave per (b, h, 16-row S tile). Scores (16 x 576) fp32 in LDS, softmax,
// un-normalized bf16 probs in LDS, PV via WMMA, 1/rowsum folded into epilogue.
__global__ __launch_bounds__(32) void attn_kernel(
    const unsigned short* __restrict__ Q,    // (B*S, HID) bf16
    const unsigned short* __restrict__ Kb,   // (B*NQ, HID) bf16
    const unsigned short* __restrict__ Vt,   // (B, H, HD, NQ) bf16
    const float* __restrict__ mask,          // (B, 1, S, NQ) fp32
    unsigned short* __restrict__ Ctx) {      // (B*S, HID) bf16
  __shared__ __align__(16) float          s_scores[16 * NQ_];
  __shared__ __align__(16) unsigned short s_probs[16 * NQ_];
  __shared__ float                        s_rsum[16];

  const int lane = threadIdx.x;
  const int s0 = blockIdx.x * 16;
  const int h  = blockIdx.y;
  const int b  = blockIdx.z;
  const int row = lane & 15, hi = lane >> 4;
  const float scale = 0.08838834764831845f;  // 1/sqrt(128)

  // ---- load Q fragments: 4 x (16x32) covering d = 0..127
  ABFrag qa[4];
  const unsigned short* qp =
      Q + (size_t)(b * S_ + s0 + row) * HID_ + (size_t)h * HD_ + hi * 8;
  for (int j = 0; j < 4; ++j) {
    qa[j].q[0] = *reinterpret_cast<const uint4*>(qp + j * 32);
    qa[j].q[1] = *reinterpret_cast<const uint4*>(qp + j * 32 + 16);
  }

  // ---- scores = Q K^T * scale + mask  (36 chunks of 16 nq)
  const unsigned short* kbase =
      Kb + (size_t)b * NQ_ * HID_ + (size_t)h * HD_ + hi * 8;
  const float* mbase = mask + ((size_t)b * S_ + s0) * NQ_;
  for (int c = 0; c < NQ_ / 16; ++c) {
    CFrag acc;
    for (int i = 0; i < 8; ++i) acc.f[i] = 0.0f;
    const unsigned short* kp = kbase + (size_t)(c * 16 + row) * HID_;
    for (int j = 0; j < 4; ++j) {
      ABFrag kf;
      kf.q[0] = *reinterpret_cast<const uint4*>(kp + j * 32);
      kf.q[1] = *reinterpret_cast<const uint4*>(kp + j * 32 + 16);
      acc.v = __builtin_amdgcn_wmma_f32_16x16x32_bf16(false, qa[j].v, false, kf.v, (short)0, acc.v, false, false);
    }
    const int n = c * 16 + row;
    for (int i = 0; i < 8; ++i) {
      const int m = i + 8 * hi;
      s_scores[m * NQ_ + n] = acc.f[i] * scale + mbase[(size_t)m * NQ_ + n];
    }
  }
  __syncthreads();

  // ---- softmax: lane pair (r, r+16) handles row r, 288 elems each half
  {
    const int base = row * NQ_ + hi * (NQ_ / 2);
    float mx = -INFINITY;
    for (int t = 0; t < NQ_ / 2; ++t) mx = fmaxf(mx, s_scores[base + t]);
    mx = fmaxf(mx, __shfl_xor(mx, 16));
    float sum = 0.0f;
    for (int t = 0; t < NQ_ / 2; ++t) {
      float e = __expf(s_scores[base + t] - mx);
      sum += e;
      s_probs[base + t] = f2bf(e);   // un-normalized; fold 1/sum into epilogue
    }
    sum += __shfl_xor(sum, 16);
    if (lane < 16) s_rsum[lane] = sum;
  }
  __syncthreads();

  float rinv[8];
  for (int i = 0; i < 8; ++i) rinv[i] = 1.0f / s_rsum[i + 8 * hi];

  // ---- out = P @ V : 8 d-chunks of 16, 18 q-chunks of 32
  const unsigned short* vtb = Vt + ((size_t)b * H_ + h) * HD_ * NQ_;
  unsigned short* cp = Ctx + (size_t)(b * S_ + s0) * HID_ + (size_t)h * HD_;
  for (int j = 0; j < HD_ / 16; ++j) {
    CFrag acc;
    for (int i = 0; i < 8; ++i) acc.f[i] = 0.0f;
    for (int qc = 0; qc < NQ_ / 32; ++qc) {
      ABFrag pa, vb;
      const unsigned short* pp = s_probs + row * NQ_ + qc * 32 + hi * 8;
      pa.q[0] = *reinterpret_cast<const uint4*>(pp);
      pa.q[1] = *reinterpret_cast<const uint4*>(pp + 16);
      const unsigned short* vp =
          vtb + (size_t)(j * 16 + row) * NQ_ + qc * 32 + hi * 8;
      vb.q[0] = *reinterpret_cast<const uint4*>(vp);
      vb.q[1] = *reinterpret_cast<const uint4*>(vp + 16);
      acc.v = __builtin_amdgcn_wmma_f32_16x16x32_bf16(false, pa.v, false, vb.v, (short)0, acc.v, false, false);
    }
    for (int i = 0; i < 8; ++i) {
      const int m = i + 8 * hi;
      cp[(size_t)m * HID_ + j * 16 + row] = f2bf(acc.f[i] * rinv[i]);
    }
  }
}

// ---------------------------------------------------------------- launch
extern "C" void kernel_launch(void* const* d_in, const int* in_sizes, int n_in,
                              void* d_out, int out_size, void* d_ws, size_t ws_size,
                              hipStream_t stream) {
  const float* hs   = (const float*)d_in[0];  // (B,S,HID)
  const float* bp   = (const float*)d_in[1];  // (B,NQ,HID)
  const float* mask = (const float*)d_in[2];  // (B,1,S,NQ)
  const float* Wq   = (const float*)d_in[3];
  const float* Wk   = (const float*)d_in[4];
  const float* Wv   = (const float*)d_in[5];
  const float* Wo   = (const float*)d_in[6];
  float* out = (float*)d_out;

  const size_t szHS = (size_t)B_ * S_ * HID_;    // 33,554,432
  const size_t szBP = (size_t)B_ * NQ_ * HID_;   //  4,718,592
  const size_t szW  = (size_t)HID_ * HID_;       // 16,777,216
  const size_t szVT = (size_t)B_ * H_ * HD_ * NQ_;

  unsigned short* p = (unsigned short*)d_ws;
  unsigned short* hsb = p;  p += szHS;
  unsigned short* bpb = p;  p += szBP;
  unsigned short* wqb = p;  p += szW;
  unsigned short* wkb = p;  p += szW;
  unsigned short* wvb = p;  p += szW;
  unsigned short* wob = p;  p += szW;
  unsigned short* qb  = p;  p += szHS;
  unsigned short* kb  = p;  p += szBP;
  unsigned short* vb  = p;  p += szBP;
  unsigned short* vtb = p;  p += szVT;
  unsigned short* ctx = p;  p += szHS;

  // 1) fp32 -> bf16 (vectorized: 8 elems/thread)
  cvt_f32_bf16_v8<<<2048, 256, 0, stream>>>((const float4*)hs, (uint4*)hsb, szHS / 8);
  cvt_f32_bf16_v8<<<2048, 256, 0, stream>>>((const float4*)bp, (uint4*)bpb, szBP / 8);
  cvt_f32_bf16_v8<<<2048, 256, 0, stream>>>((const float4*)Wq, (uint4*)wqb, szW / 8);
  cvt_f32_bf16_v8<<<2048, 256, 0, stream>>>((const float4*)Wk, (uint4*)wkb, szW / 8);
  cvt_f32_bf16_v8<<<2048, 256, 0, stream>>>((const float4*)Wv, (uint4*)wvb, szW / 8);
  cvt_f32_bf16_v8<<<2048, 256, 0, stream>>>((const float4*)Wo, (uint4*)wob, szW / 8);

  // 2) projections  (block tile 64M x 256N)
  dim3 gq(HID_ / 256, (B_ * S_) / 64);   // (16, 128)
  dim3 gkv(HID_ / 256, (B_ * NQ_) / 64); // (16, 18)
  gemm_xwt<true><<<gq,  256, 0, stream>>>(hsb, wqb, qb, B_ * S_,  HID_, HID_);
  gemm_xwt<true><<<gkv, 256, 0, stream>>>(bpb, wkb, kb, B_ * NQ_, HID_, HID_);
  gemm_xwt<true><<<gkv, 256, 0, stream>>>(bpb, wvb, vb, B_ * NQ_, HID_, HID_);

  // 3) V transpose for contiguous PV B-fragments
  transpose_v_kernel<<<(int)((szVT + 255) / 256), 256, 0, stream>>>(vb, vtb);

  // 4) attention (flash-style, 16-row S tiles)
  dim3 ga(S_ / 16, H_, B_);              // (256, 32, 2)
  attn_kernel<<<ga, 32, 0, stream>>>(qb, kb, vtb, mask, ctx);

  // 5) output projection -> fp32
  gemm_xwt<false><<<gq, 256, 0, stream>>>(ctx, wob, out, B_ * S_, HID_, HID_);
}
